// UResidualMambaSeq_68564857913720
// MI455X (gfx1250) — matrix operand order repaired and checked by reading, not compiled, not run on hardware
//
#include <hip/hip_runtime.h>
#include <hip/hip_bf16.h>

typedef __attribute__((ext_vector_type(16))) _Float16 v16h;
typedef __attribute__((ext_vector_type(8)))  _Float16 v8h;
typedef __attribute__((ext_vector_type(4)))  _Float16 v4h;
typedef __attribute__((ext_vector_type(8)))  float    v8f;
typedef __attribute__((ext_vector_type(4)))  float    v4f;

#define BT 4096      // B * T rows in every GEMM
#define TLEN 512
#define NB 8
#define NSTATE 16

// ---------------------------------------------------------------------------
// f32 -> f16 pack (4 elements / thread, 128-bit load, 64-bit store)
// n must be a multiple of 4 (all sizes here are powers-of-two products)
// ---------------------------------------------------------------------------
__global__ void f32_to_f16_kernel(const float* __restrict__ s,
                                  _Float16* __restrict__ d, int n4) {
  int i = blockIdx.x * blockDim.x + threadIdx.x;
  if (i >= n4) return;
  v4f v = ((const v4f*)s)[i];
  v4h o;
#pragma unroll
  for (int j = 0; j < 4; ++j) o[j] = (_Float16)v[j];
  ((v4h*)d)[i] = o;
}

// ---------------------------------------------------------------------------
// WMMA GEMM:  Y[M,N] = X[M,K] @ W[N,K]^T (+ bias[N]),  X/W pre-packed f16.
// Requires K % 64 == 0 (guaranteed by caller). One wave per 16x16 tile.
// grid = (ceil(N/16), M/16), block = 32.
//
// Fragment layouts (CDNA5 ISA):
//   A 16x32 f16 : lane L -> row M=L&15, elem j -> K = 8*(L>=16) + (j<8?j:j+8)
//                 => two contiguous 8-half runs at offsets half*8 and half*8+16
//   B 32x16 f16 : lane L -> col N=L&15, elem j -> K = 8*(L>=16) + (j<8?j:j+8)
//                 => two contiguous 8-half runs at offsets half*8 and half*8+16
//   C/D f32     : lane L -> col N=L&15, vgpr r -> row M = r + 8*(L>=16)
// ---------------------------------------------------------------------------
__device__ __forceinline__ v16h load_a_frag(const _Float16* __restrict__ p) {
  v8h lo = *(const v8h*)(p);        // K = base + 0..7
  v8h hi = *(const v8h*)(p + 16);   // K = base + 16..23
  v16h a;
#pragma unroll
  for (int j = 0; j < 8; ++j) { a[j] = lo[j]; a[8 + j] = hi[j]; }
  return a;
}

__device__ __forceinline__ v16h load_b_frag(const _Float16* __restrict__ p) {
  v8h lo = *(const v8h*)(p);
  v8h hi = *(const v8h*)(p + 16);
  v16h b;
#pragma unroll
  for (int j = 0; j < 8; ++j) { b[j] = lo[j]; b[8 + j] = hi[j]; }
  return b;
}

__global__ __launch_bounds__(32)
void gemm_wmma_f16(const _Float16* __restrict__ X,
                   const _Float16* __restrict__ W,
                   const float* __restrict__ bias,
                   float* __restrict__ Y,
                   int N, int K) {
  const int lane = threadIdx.x;        // 0..31 (wave32)
  const int half = lane >> 4;          // 0 | 1
  const int l    = lane & 15;
  const int mrow = blockIdx.y * 16 + l;
  const int ncol = blockIdx.x * 16 + l;
  const int ncl  = (ncol < N) ? ncol : (N - 1);   // clamp: junk cols never stored

  const _Float16* __restrict__ xrow = X + (size_t)mrow * K + half * 8;
  const _Float16* __restrict__ wrow = W + (size_t)ncl  * K + half * 8;

  v8f acc0 = {}, acc1 = {};
  for (int k0 = 0; k0 + 64 <= K; k0 += 64) {
    v16h a0 = load_a_frag(xrow + k0);
    v16h b0 = load_b_frag(wrow + k0);
    v16h a1 = load_a_frag(xrow + k0 + 32);
    v16h b1 = load_b_frag(wrow + k0 + 32);
    acc0 = __builtin_amdgcn_wmma_f32_16x16x32_f16(false, a0, false, b0,
                                                  (short)0, acc0, false, false);
    acc1 = __builtin_amdgcn_wmma_f32_16x16x32_f16(false, a1, false, b1,
                                                  (short)0, acc1, false, false);
  }

  if (ncol < N) {
    float bs = bias ? bias[ncol] : 0.0f;
#pragma unroll
    for (int r = 0; r < 8; ++r) {
      int mo = blockIdx.y * 16 + r + half * 8;
      Y[(size_t)mo * N + ncol] = acc0[r] + acc1[r] + bs;
    }
  }
}

// ---------------------------------------------------------------------------
// Weight-norm prep for conv1x1: w[r,c] = g[r] * v[r,c] / ||v[r,:]||
// ---------------------------------------------------------------------------
__global__ void wnorm_kernel(const float* __restrict__ v,
                             const float* __restrict__ g,
                             float* __restrict__ w, int cin) {
  __shared__ float red[256];
  const int r = blockIdx.x;
  float s = 0.0f;
  for (int c = threadIdx.x; c < cin; c += 256) {
    float t = v[(size_t)r * cin + c];
    s += t * t;
  }
  red[threadIdx.x] = s;
  __syncthreads();
  for (int st = 128; st > 0; st >>= 1) {
    if (threadIdx.x < st) red[threadIdx.x] += red[threadIdx.x + st];
    __syncthreads();
  }
  float scale = g[r] * rsqrtf(red[0]);
  for (int c = threadIdx.x; c < cin; c += 256)
    w[(size_t)r * cin + c] = v[(size_t)r * cin + c] * scale;
}

__device__ __forceinline__ float silu_f(float x) {
  return x / (1.0f + __expf(-x));
}

// ---------------------------------------------------------------------------
// Depthwise causal conv (4 taps) + bias + silu.
// ---------------------------------------------------------------------------
__global__ void conv_silu_kernel(const float* __restrict__ xz, int ld,
                                 const float* __restrict__ cw,
                                 const float* __restrict__ cb,
                                 float* __restrict__ xs_out,
                                 int din, int total) {
  int i = blockIdx.x * blockDim.x + threadIdx.x;
  if (i >= total) return;
  int c = i % din;
  int m = i / din;
  int t = m % TLEN;
  int mb = m - t;
  float s = 0.0f;
#pragma unroll
  for (int k = 0; k < 4; ++k) {
    int tt = t + k - 3;
    if (tt >= 0) s += xz[(size_t)(mb + tt) * ld + c] * cw[c * 4 + k];
  }
  s += cb[c];
  xs_out[i] = silu_f(s);
}

// ---------------------------------------------------------------------------
// dt projection (tiny K = dr <= 16): dt_raw = dbl[:, :dr] @ dt_w.T + dt_b
// ---------------------------------------------------------------------------
__global__ void dtproj_kernel(const float* __restrict__ dbl, int ldd, int dr,
                              const float* __restrict__ dt_w,
                              const float* __restrict__ dt_b,
                              float* __restrict__ dt, int din, int total) {
  int i = blockIdx.x * blockDim.x + threadIdx.x;
  if (i >= total) return;
  int c = i % din;
  int m = i / din;
  const float* __restrict__ row = dbl + (size_t)m * ldd;
  const float* __restrict__ w   = dt_w + (size_t)c * dr;
  float s = dt_b[c];
  for (int k = 0; k < dr; ++k) s += row[k] * w[k];
  dt[i] = s;
}

// ---------------------------------------------------------------------------
// Selective scan: grid = (ceil(din/256), NB). One thread per (batch, channel),
// 16-wide state in regs; per-(b,t) B/C row (32 f32) staged through LDS.
// ---------------------------------------------------------------------------
__global__ void scan_kernel(const float* __restrict__ dtraw,
                            const float* __restrict__ dbl, int ldd, int dr,
                            const float* __restrict__ xs,
                            const float* __restrict__ A_log,
                            float* __restrict__ y, int din) {
  __shared__ float sBC[32];
  const int b = blockIdx.y;
  const int c = blockIdx.x * blockDim.x + threadIdx.x;
  const bool active = (c < din);

  float A[NSTATE], h[NSTATE];
  if (active) {
#pragma unroll
    for (int n = 0; n < NSTATE; ++n) {
      A[n] = -__expf(A_log[c * NSTATE + n]);
      h[n] = 0.0f;
    }
  }
  for (int t = 0; t < TLEN; ++t) {
    const int m = b * TLEN + t;
    __syncthreads();
    if (threadIdx.x < 32)
      sBC[threadIdx.x] = dbl[(size_t)m * ldd + dr + threadIdx.x];
    __syncthreads();
    if (active) {
      float dt = dtraw[(size_t)m * din + c];
      dt = (dt > 20.0f) ? dt : log1pf(__expf(dt));      // softplus
      float dx = dt * xs[(size_t)m * din + c];
      float acc = 0.0f;
#pragma unroll
      for (int n = 0; n < NSTATE; ++n) {
        h[n] = __expf(dt * A[n]) * h[n] + dx * sBC[n];
        acc += h[n] * sBC[NSTATE + n];
      }
      y[(size_t)m * din + c] = acc;
    }
  }
}

// ---------------------------------------------------------------------------
// Gate: y = (y + xs*D[c]) * silu(z);  z = xz[:, din + c]
// ---------------------------------------------------------------------------
__global__ void gate_kernel(float* __restrict__ y,
                            const float* __restrict__ xs,
                            const float* __restrict__ Dp,
                            const float* __restrict__ xz, int ld,
                            int din, int total) {
  int i = blockIdx.x * blockDim.x + threadIdx.x;
  if (i >= total) return;
  int c = i % din;
  int m = i / din;
  float z = xz[(size_t)m * ld + din + c];
  y[i] = (y[i] + xs[i] * Dp[c]) * silu_f(z);
}

__global__ void add_inplace_kernel(float* __restrict__ x,
                                   const float* __restrict__ r, int n) {
  int i = blockIdx.x * blockDim.x + threadIdx.x;
  if (i < n) x[i] += r[i];
}

__global__ void add_out_kernel(const float* __restrict__ a,
                               const float* __restrict__ b,
                               float* __restrict__ o, int n) {
  int i = blockIdx.x * blockDim.x + threadIdx.x;
  if (i < n) o[i] = a[i] + b[i];
}

__global__ void copy_kernel(float* __restrict__ d,
                            const float* __restrict__ s, int n) {
  int i = blockIdx.x * blockDim.x + threadIdx.x;
  if (i < n) d[i] = s[i];
}

// ---------------------------------------------------------------------------
// Host orchestration
// ---------------------------------------------------------------------------
extern "C" void kernel_launch(void* const* d_in, const int* in_sizes, int n_in,
                              void* d_out, int out_size, void* d_ws, size_t ws_size,
                              hipStream_t stream) {
  (void)in_sizes; (void)n_in; (void)out_size; (void)ws_size;

  int idx = 0;
  auto nxt = [&]() { return (const float*)d_in[idx++]; };

  struct ConvP { const float *v, *g, *b; };
  struct MB    { const float* p[9]; };   // in_w,conv_w,conv_b,xproj_w,dt_w,dt_b,A_log,D,out_w

  const float* xin = nxt();
  ConvP dc[2]; MB dm[2][2]; MB mm[2][2]; ConvP uc[2]; MB um[2][2];
  for (int i = 0; i < 2; ++i) { dc[i].v = nxt(); dc[i].g = nxt(); dc[i].b = nxt(); }
  for (int i = 0; i < 2; ++i)
    for (int j = 0; j < 2; ++j)
      for (int k = 0; k < 9; ++k) dm[i][j].p[k] = nxt();
  for (int i = 0; i < 2; ++i)
    for (int j = 0; j < 2; ++j)
      for (int k = 0; k < 9; ++k) mm[i][j].p[k] = nxt();
  for (int i = 0; i < 2; ++i) { uc[i].v = nxt(); uc[i].g = nxt(); uc[i].b = nxt(); }
  for (int i = 0; i < 2; ++i)
    for (int j = 0; j < 2; ++j)
      for (int k = 0; k < 9; ++k) um[i][j].p[k] = nxt();

  // workspace arena (float granularity; all offsets 16B-aligned)
  float* base = (float*)d_ws;
  size_t off = 0;
  auto alloc = [&](size_t n) { float* p = base + off; off += n; return p; };
  float* xA      = alloc((size_t)BT * 256);
  float* xB      = alloc((size_t)BT * 256);
  float* res0    = alloc((size_t)BT * 256);
  float* res1    = alloc((size_t)BT * 128);
  float* buf_xz  = alloc((size_t)BT * 1024);
  float* buf_xs  = alloc((size_t)BT * 512);
  float* buf_dbl = alloc((size_t)BT * 48);
  float* buf_dt  = alloc((size_t)BT * 512);
  float* buf_y   = alloc((size_t)BT * 512);
  float* buf_w   = alloc((size_t)64 * 1024);
  _Float16* hX   = (_Float16*)alloc((size_t)BT * 256);   // BT*512 halves
  _Float16* hW   = (_Float16*)alloc((size_t)1024 * 128); // 256K halves

  float* cur = xA;
  float* other = xB;

  auto launch1d = [&](int n) { return dim3((unsigned)((n + 255) / 256)); };
  auto cvt = [&](const float* s, _Float16* d, int n) {
    int n4 = n / 4;
    f32_to_f16_kernel<<<launch1d(n4), 256, 0, stream>>>(s, d, n4);
  };
  // Y = X[M,K] @ W[N,K]^T + bias ; X,W are f32, packed to f16 first. K%64==0.
  auto gemm = [&](const float* X, const float* W, const float* bias,
                  float* Y, int N, int K) {
    cvt(X, hX, BT * K);
    cvt(W, hW, N * K);
    dim3 g((unsigned)((N + 15) / 16), BT / 16);
    gemm_wmma_f16<<<g, 32, 0, stream>>>(hX, hW, bias, Y, N, K);
  };

  auto conv1x1 = [&](const ConvP& c, int cin, int cout) {
    wnorm_kernel<<<cout, 256, 0, stream>>>(c.v, c.g, buf_w, cin);
    gemm(cur, buf_w, c.b, other, cout, cin);
    float* t = cur; cur = other; other = t;
  };

  auto mamba = [&](const MB& m, int d) {
    const int din = 2 * d;
    const int dr  = (d + 15) / 16;
    const int ldz = 2 * din;          // xz columns
    const int nbl = dr + 2 * NSTATE;  // dbl columns
    const int tot = BT * din;
    const float* in_w    = m.p[0];
    const float* conv_w  = m.p[1];
    const float* conv_b  = m.p[2];
    const float* xproj_w = m.p[3];
    const float* dt_w    = m.p[4];
    const float* dt_b    = m.p[5];
    const float* A_log   = m.p[6];
    const float* Dp      = m.p[7];
    const float* out_w   = m.p[8];

    gemm(cur, in_w, nullptr, buf_xz, ldz, d);                          // xz
    conv_silu_kernel<<<launch1d(tot), 256, 0, stream>>>(buf_xz, ldz, conv_w, conv_b,
                                                        buf_xs, din, tot);
    gemm(buf_xs, xproj_w, nullptr, buf_dbl, nbl, din);                 // dbl
    dtproj_kernel<<<launch1d(tot), 256, 0, stream>>>(buf_dbl, nbl, dr, dt_w, dt_b,
                                                     buf_dt, din, tot);
    dim3 sg((unsigned)((din + 255) / 256), NB);
    scan_kernel<<<sg, 256, 0, stream>>>(buf_dt, buf_dbl, nbl, dr,
                                        buf_xs, A_log, buf_y, din);
    gate_kernel<<<launch1d(tot), 256, 0, stream>>>(buf_y, buf_xs, Dp, buf_xz, ldz,
                                                   din, tot);
    gemm(buf_y, out_w, nullptr, other, d, din);                        // out proj
    float* t = cur; cur = other; other = t;
  };

  // ---- forward pass ----
  copy_kernel<<<launch1d(BT * 256), 256, 0, stream>>>(cur, xin, BT * 256);

  // down 0: 256 -> 128
  copy_kernel<<<launch1d(BT * 256), 256, 0, stream>>>(res0, cur, BT * 256);
  conv1x1(dc[0], 256, 128);
  mamba(dm[0][0], 128);
  mamba(dm[0][1], 128);

  // down 1: 128 -> 64
  copy_kernel<<<launch1d(BT * 128), 256, 0, stream>>>(res1, cur, BT * 128);
  conv1x1(dc[1], 128, 64);
  mamba(dm[1][0], 64);
  mamba(dm[1][1], 64);

  // mid
  mamba(mm[0][0], 64);
  mamba(mm[0][1], 64);
  mamba(mm[1][0], 64);
  mamba(mm[1][1], 64);

  // up 0: 64 -> 128, + res1
  conv1x1(uc[0], 64, 128);
  mamba(um[0][0], 128);
  mamba(um[0][1], 128);
  add_inplace_kernel<<<launch1d(BT * 128), 256, 0, stream>>>(cur, res1, BT * 128);

  // up 1: 128 -> 256, + res0 -> d_out
  conv1x1(uc[1], 128, 256);
  mamba(um[1][0], 256);
  mamba(um[1][1], 256);
  add_out_kernel<<<launch1d(BT * 256), 256, 0, stream>>>(cur, res0, (float*)d_out,
                                                         BT * 256);
}
